// MoEPatchDecoder_71605694759010
// MI455X (gfx1250) — matrix operand order repaired and checked by reading, not compiled, not run on hardware
//
#include <hip/hip_runtime.h>
#include <hip/hip_bf16.h>
#include <hip/hip_fp16.h>

typedef __attribute__((ext_vector_type(16))) _Float16 v16h;
typedef __attribute__((ext_vector_type(8)))  _Float16 v8h;
typedef __attribute__((ext_vector_type(8)))  float    v8f;

#define HH 512
#define WW 512
#define HWSZ (HH*WW)

// ---------------------------------------------------------------------------
// Weight reorder: f32 [COUT][CIN][3][3]  ->  f16 [COUT][9][CIN]
// K ordering is tap-major so each 32-wide WMMA K-chunk is one tap + 32 cins.
// ---------------------------------------------------------------------------
__global__ void wconvert_kernel(const float* __restrict__ src,
                                _Float16* __restrict__ dst,
                                int COUT, int CIN) {
  int total = COUT * CIN * 9;
  for (int idx = blockIdx.x * blockDim.x + threadIdx.x; idx < total;
       idx += gridDim.x * blockDim.x) {
    int cout = idx / (CIN * 9);
    int rem  = idx - cout * CIN * 9;
    int cin  = rem / 9;
    int tap  = rem - cin * 9;
    dst[((size_t)cout * 9 + tap) * CIN + cin] = (_Float16)src[idx];
  }
}

// ---------------------------------------------------------------------------
// 1D bilinear x2 upsample (half-pixel centers, edge clamp), fully unrolled
// so it constant-folds.
// ---------------------------------------------------------------------------
template <int N>
__device__ __forceinline__ void up1d(const float* in, float* out) {
#pragma unroll
  for (int i = 0; i < 2 * N; ++i) {
    float s  = (i + 0.5f) * 0.5f - 0.5f;
    float fl = floorf(s);
    int   i0 = (int)fl;
    float f  = s - fl;
    int lo = i0 < 0 ? 0 : (i0 > N - 1 ? N - 1 : i0);
    int hi = (i0 + 1) < 0 ? 0 : ((i0 + 1) > N - 1 ? N - 1 : (i0 + 1));
    out[i] = (1.f - f) * in[lo] + f * in[hi];
  }
}

// ---------------------------------------------------------------------------
// Per-patch expert decode + overlapping scatter-add.
// Upsample(2x2 -> 16x16) is linear, so apply the 64x128 expert GEMM on the
// 2x2 latent first (ysh), then expand with the composite 16x2 weight matrix.
// ---------------------------------------------------------------------------
__global__ __launch_bounds__(256) void decode_scatter_kernel(
    const float* __restrict__ latents, const int* __restrict__ expert_ids,
    const int* __restrict__ positions, const float* __restrict__ dec_w,
    const float* __restrict__ dec_b, float* __restrict__ feat,
    float* __restrict__ cnt) {
  __shared__ float ysh[64][4];
  __shared__ float Msh[16][2];

  const int n   = blockIdx.x;
  const int tid = threadIdx.x;
  const int e   = expert_ids[n];

  if (tid == 0) {
    float c2a[2] = {1.f, 0.f}, c2b[2] = {0.f, 1.f};
    float a4[4], a8[8], a16[16], b4a[4], b8[8], b16[16];
    up1d<2>(c2a, a4);  up1d<4>(a4, a8);   up1d<8>(a8, a16);
    up1d<2>(c2b, b4a); up1d<4>(b4a, b8);  up1d<8>(b8, b16);
#pragma unroll
    for (int i = 0; i < 16; ++i) { Msh[i][0] = a16[i]; Msh[i][1] = b16[i]; }
  }

  { // phase 1: y[o][p] = sum_c dec_w[e][o][c] * lat[n][c][p]
    const int o = tid >> 2, p = tid & 3;
    const float* wrow = dec_w + ((size_t)e * 64 + o) * 128;
    const float* lrow = latents + (size_t)n * 512 + p;
    float acc = 0.f;
#pragma unroll 4
    for (int c = 0; c < 128; ++c) acc += wrow[c] * lrow[c * 4];
    ysh[o][p] = acc;
  }
  __syncthreads();

  // phase 2: one thread per output pixel (i,j) of the 16x16 patch
  const int i = tid >> 4, j = tid & 15;
  const int posx = positions[n * 2 + 0];
  const int posy = positions[n * 2 + 1];
  const int gy = posy * 16 + i, gx = posx * 16 + j;
  const int pix = gy * WW + gx;
  const float mi0 = Msh[i][0], mi1 = Msh[i][1];
  const float mj0 = Msh[j][0], mj1 = Msh[j][1];
  const float* bias = dec_b + (size_t)e * 64;
  for (int o = 0; o < 64; ++o) {
    float v = mi0 * (mj0 * ysh[o][0] + mj1 * ysh[o][1]) +
              mi1 * (mj0 * ysh[o][2] + mj1 * ysh[o][3]) + bias[o];
    atomicAdd(&feat[(size_t)o * HWSZ + pix], v);
  }
  atomicAdd(&cnt[pix], 1.0f);
}

// ---------------------------------------------------------------------------
// 3x3 SAME conv as implicit GEMM on v_wmma_f32_16x16x32_f16 (wave32).
// Block = 256 threads (8 waves), 16x16 pixel tile; wave w owns rows 2w,2w+1.
// LDS tile [y][x][c] with per-pixel stride padded 64 -> 72 halfs so the 16
// lanes of a ds_load_b128 hit all sixteen 4-bank windows (64-bank LDS)
// instead of a 16-way pile-up on 2 windows.
// NHWC input is staged with GLOBAL_LOAD_ASYNC_TO_LDS_B128 (ASYNCcnt path);
// IN_PLANAR (conv1) reads planar f32 feat / max(cnt,1) via VGPRs.
// ---------------------------------------------------------------------------
template <int CIN, int COUT, bool RELU, bool IN_PLANAR>
__global__ __launch_bounds__(256) void conv3x3_wmma_kernel(
    const void* __restrict__ srcv, const float* __restrict__ cnt,
    const _Float16* __restrict__ wT, const float* __restrict__ bias,
    _Float16* __restrict__ dst) {
  constexpr int CCH = 64;           // channel chunk staged in LDS
  constexpr int STR = CCH + 8;      // padded per-pixel stride (halfs)
  constexpr int NCH = CIN / CCH;
  __shared__ _Float16 smem[18 * 18 * STR];

  const int tid  = threadIdx.x;
  const int lane = tid & 31;
  const int wv   = tid >> 5;        // wave 0..7
  const int px   = lane & 15;       // N index (pixel column) / M index (cout)
  const int hi   = lane >> 4;       // K-half select (this code's contiguous k = 16*hi + s order)
  const int tx0  = blockIdx.x * 16;
  const int ty0  = blockIdx.y * 16;

  for (int cb = 0; cb < COUT / 64; ++cb) {
    v8f acc[4][2];
#pragma unroll
    for (int t = 0; t < 4; ++t)
#pragma unroll
      for (int rr = 0; rr < 2; ++rr)
#pragma unroll
        for (int r = 0; r < 8; ++r) acc[t][rr][r] = 0.f;

    for (int ch = 0; ch < NCH; ++ch) {
      const int c0 = ch * CCH;
      __syncthreads();
      // ---- stage 18x18 halo tile, channels [c0, c0+CCH), into LDS ----
      for (int s = tid; s < 18 * 18; s += 256) {
        const int ly = s / 18, lx = s - ly * 18;
        const int gy = ty0 + ly - 1, gx = tx0 + lx - 1;
        const bool inb = (gy >= 0) && (gy < HH) && (gx >= 0) && (gx < WW);
        _Float16* dptr = &smem[s * STR];
        if (IN_PLANAR) {
          const float* sp = (const float*)srcv;
          float inv = 1.f;
          int pbase = 0;
          if (inb) {
            pbase = gy * WW + gx;
            inv   = 1.0f / fmaxf(cnt[pbase], 1.f);
          }
#pragma unroll 8
          for (int c = 0; c < CCH; ++c) {
            float v = inb ? sp[(size_t)(c0 + c) * HWSZ + pbase] * inv : 0.f;
            dptr[c] = (_Float16)v;
          }
        } else {
          if (inb) {
            const _Float16* gp0 = (const _Float16*)srcv +
                                  ((long long)(gy * WW + gx) * CIN + c0);
#pragma unroll
            for (int c = 0; c < CCH; c += 8) {
              // async DMA: global (GV 64-bit vaddr) -> LDS (byte offset VGPR)
              const unsigned lds_off =
                  (unsigned)(unsigned long long)(const void*)(dptr + c);
              const void* gp = (const void*)(gp0 + c);
              asm volatile("global_load_async_to_lds_b128 %0, %1, off"
                           :
                           : "v"(lds_off), "v"(gp)
                           : "memory");
            }
          } else {
            v8h z;
#pragma unroll
            for (int e = 0; e < 8; ++e) z[e] = (_Float16)0.f;
#pragma unroll
            for (int c = 0; c < CCH; c += 8) *(v8h*)(dptr + c) = z;
          }
        }
      }
      if (!IN_PLANAR) {
        asm volatile("s_wait_asynccnt 0x0" ::: "memory");
      }
      __syncthreads();

      // ---- K loop: 9 taps x (CCH/32) chunks of 32 channels ----
#pragma unroll
      for (int tap = 0; tap < 9; ++tap) {
        const int ky = tap / 3, kx = tap - ky * 3;
#pragma unroll
        for (int cc = 0; cc < CCH; cc += 32) {
          // B fragments for the wave's two pixel rows (16-B aligned pairs)
          const int b0 = ((2 * wv + ky) * 18 + (px + kx)) * STR + cc + hi * 16;
          const v8h b0lo = *(const v8h*)&smem[b0];
          const v8h b0hi = *(const v8h*)&smem[b0 + 8];
          const v8h b1lo = *(const v8h*)&smem[b0 + 18 * STR];
          const v8h b1hi = *(const v8h*)&smem[b0 + 18 * STR + 8];
          v16h B0, B1;
#pragma unroll
          for (int e = 0; e < 8; ++e) {
            B0[e] = b0lo[e]; B0[e + 8] = b0hi[e];
            B1[e] = b1lo[e]; B1[e + 8] = b1hi[e];
          }
#pragma unroll
          for (int t = 0; t < 4; ++t) {
            const int cout = cb * 64 + t * 16 + px;   // M = lane % 16
            const _Float16* wp =
                wT + ((size_t)cout * 9 + tap) * CIN + c0 + cc + hi * 16;
            const v8h a0 = *(const v8h*)wp;           // K elems 0..7  of half
            const v8h a1 = *(const v8h*)(wp + 8);     // K elems 8..15 of half
            v16h A;
#pragma unroll
            for (int e = 0; e < 8; ++e) { A[e] = a0[e]; A[e + 8] = a1[e]; }
            acc[t][0] = __builtin_amdgcn_wmma_f32_16x16x32_f16(
                false, A, false, B0, (short)0, acc[t][0], false, false);
            acc[t][1] = __builtin_amdgcn_wmma_f32_16x16x32_f16(
                false, A, false, B1, (short)0, acc[t][1], false, false);
          }
        }
      }
    }

    // ---- epilogue: bias + ReLU, pack 8 consecutive channels, NHWC store ----
#pragma unroll
    for (int t = 0; t < 4; ++t) {
      const int chb = cb * 64 + t * 16 + hi * 8;      // M = r + hi*8 layout
#pragma unroll
      for (int rr = 0; rr < 2; ++rr) {
        const int gy = ty0 + 2 * wv + rr;
        const int gx = tx0 + px;
        v8h outv;
#pragma unroll
        for (int r = 0; r < 8; ++r) {
          float v = acc[t][rr][r] + bias[chb + r];
          if (RELU) v = fmaxf(v, 0.f);
          outv[r] = (_Float16)v;
        }
        *(v8h*)&dst[((size_t)(gy * WW + gx)) * COUT + chb] = outv;
      }
    }
  }
}

// ---------------------------------------------------------------------------
// Final 3x3 conv, 128 -> 3 channels (no ReLU), f32 NCHW output. Too skinny
// for WMMA tiles; VALU with weights cached in LDS (uniform-broadcast reads).
// ---------------------------------------------------------------------------
__global__ __launch_bounds__(256) void conv4_kernel(
    const _Float16* __restrict__ act, const float* __restrict__ w4,
    const float* __restrict__ b4, float* __restrict__ out) {
  __shared__ float wsh[3 * 128 * 9];
  for (int i = threadIdx.x; i < 3 * 128 * 9; i += 256) wsh[i] = w4[i];
  __syncthreads();

  const int pix = blockIdx.x * 256 + threadIdx.x;
  const int gy = pix >> 9, gx = pix & 511;
  float a0 = b4[0], a1 = b4[1], a2 = b4[2];
#pragma unroll
  for (int ky = 0; ky < 3; ++ky) {
    const int iy = gy + ky - 1;
    if (iy < 0 || iy >= HH) continue;
#pragma unroll
    for (int kx = 0; kx < 3; ++kx) {
      const int ix = gx + kx - 1;
      if (ix < 0 || ix >= WW) continue;
      const _Float16* ap = act + ((size_t)(iy * WW + ix)) * 128;
      const int wo = ky * 3 + kx;
      for (int c = 0; c < 128; ++c) {
        const float av = (float)ap[c];
        a0 += av * wsh[(0 * 128 + c) * 9 + wo];
        a1 += av * wsh[(1 * 128 + c) * 9 + wo];
        a2 += av * wsh[(2 * 128 + c) * 9 + wo];
      }
    }
  }
  out[0 * (size_t)HWSZ + pix] = a0;
  out[1 * (size_t)HWSZ + pix] = a1;
  out[2 * (size_t)HWSZ + pix] = a2;
}

// ---------------------------------------------------------------------------
// Workspace layout (bytes):
//   [0, 64Mi)        feat f32 [64][HW]   -> later aliased as act3 f16 [HW][128]
//   [64Mi, 65Mi)     cnt f32 [HW]
//   [65Mi, 129Mi)    act1 f16 [HW][128]
//   [129Mi, 257Mi)   act2 f16 [HW][256]
//   [257Mi, ...)     f16 weights w1h/w2h/w3h (~1.3 MB)
// ---------------------------------------------------------------------------
extern "C" void kernel_launch(void* const* d_in, const int* in_sizes, int n_in,
                              void* d_out, int out_size, void* d_ws,
                              size_t ws_size, hipStream_t stream) {
  (void)in_sizes; (void)n_in; (void)out_size; (void)ws_size;

  const float* latents    = (const float*)d_in[0];
  const int*   expert_ids = (const int*)d_in[1];
  const int*   positions  = (const int*)d_in[2];
  const float* dec_w      = (const float*)d_in[3];
  const float* dec_b      = (const float*)d_in[4];
  const float* w1 = (const float*)d_in[5];
  const float* b1 = (const float*)d_in[6];
  const float* w2 = (const float*)d_in[7];
  const float* b2 = (const float*)d_in[8];
  const float* w3 = (const float*)d_in[9];
  const float* b3 = (const float*)d_in[10];
  const float* w4 = (const float*)d_in[11];
  const float* b4 = (const float*)d_in[12];

  char* ws = (char*)d_ws;
  float*    feat = (float*)ws;
  float*    cnt  = (float*)(ws + (size_t)67108864);
  _Float16* act1 = (_Float16*)(ws + (size_t)68157440);
  _Float16* act2 = (_Float16*)(ws + (size_t)135266304);
  _Float16* act3 = (_Float16*)ws;  // aliases feat (feat dead after conv1)
  _Float16* w1h  = (_Float16*)(ws + (size_t)269484032);
  _Float16* w2h  = (_Float16*)(ws + (size_t)269484032 + 147456);
  _Float16* w3h  = (_Float16*)(ws + (size_t)269484032 + 147456 + 589824);

  (void)hipMemsetAsync(ws, 0, (size_t)68157440, stream);  // feat + cnt

  wconvert_kernel<<<256, 256, 0, stream>>>(w1, w1h, 128, 64);
  wconvert_kernel<<<256, 256, 0, stream>>>(w2, w2h, 256, 128);
  wconvert_kernel<<<256, 256, 0, stream>>>(w3, w3h, 128, 256);

  decode_scatter_kernel<<<1024, 256, 0, stream>>>(latents, expert_ids,
                                                  positions, dec_w, dec_b,
                                                  feat, cnt);
  dim3 g(32, 32);
  conv3x3_wmma_kernel<64, 128, true, true>
      <<<g, 256, 0, stream>>>(feat, cnt, w1h, b1, act1);
  conv3x3_wmma_kernel<128, 256, true, false>
      <<<g, 256, 0, stream>>>(act1, nullptr, w2h, b2, act2);
  conv3x3_wmma_kernel<256, 128, true, false>
      <<<g, 256, 0, stream>>>(act2, nullptr, w3h, b3, act3);
  conv4_kernel<<<1024, 256, 0, stream>>>(act3, w4, b4, (float*)d_out);
}
